// EncoderLayer_13752485282255
// MI455X (gfx1250) — matrix operand order repaired and checked by reading, not compiled, not run on hardware
//
#include <hip/hip_runtime.h>

// ---------------------------------------------------------------------------
// CDNA5 (gfx1250) transformer encoder layer, bf16 WMMA everywhere.
// Round 3: fix async-builtin pointer types (AS1/AS3 int4*, non-const) per
//          hipcc diagnostic; async global->LDS double-buffered GEMM,
//          K-step 64, prefetch hints in flash attention.
// ---------------------------------------------------------------------------

typedef __bf16 bf16_t;
typedef __attribute__((ext_vector_type(16))) __bf16 v16bf;
typedef __attribute__((ext_vector_type(8)))  __bf16 v8bf;
typedef __attribute__((ext_vector_type(8)))  float  v8f;

#define D_MODEL 1024
#define N_HEADS 16
#define DK      64
#define D_FF    4096
#define BATCH   2
#define SEQ     2048
#define TOKENS  (BATCH * SEQ)

// ---- gfx1250 async global->LDS path (guarded: falls back to reg staging) ---
#if defined(__has_builtin)
#if __has_builtin(__builtin_amdgcn_global_load_async_to_lds_b128) && \
    __has_builtin(__builtin_amdgcn_s_wait_asynccnt)
#define USE_ASYNC_LDS 1
#endif
#endif
#ifndef USE_ASYNC_LDS
#define USE_ASYNC_LDS 0
#endif

#if USE_ASYNC_LDS
typedef int v4i __attribute__((ext_vector_type(4)));
typedef __attribute__((address_space(1))) v4i* g_v4i_p;   // global int4*
typedef __attribute__((address_space(3))) v4i* l_v4i_p;   // LDS int4*

__device__ __forceinline__ void async_copy_b128(const bf16_t* gsrc, bf16_t* ldst) {
  __builtin_amdgcn_global_load_async_to_lds_b128(
      (g_v4i_p)(void*)gsrc, (l_v4i_p)(void*)ldst, 0, 0);
}
#endif

// ---------------------------------------------------------------------------
// Fragment loaders (layouts per CDNA5 ISA §7.12.2, wave32).
// A 16x32 bf16: row = lane&15; lanes<16 kb=0, lanes>=16 kb=8;
//   VGPR0..3 = K kb..kb+7, VGPR4..7 = K kb+16..kb+23  -> two 16B chunks.
// B 32x16 bf16 (stored transposed, [N][K]): n = lane&15;
//   same K order as A: lanes<16 kb=0, lanes>=16 kb=8; K kb..kb+7, kb+16..kb+23.
// ---------------------------------------------------------------------------
__device__ __forceinline__ v16bf load_a_frag(const bf16_t* base, int stride) {
  const int lane = threadIdx.x & 31;
  const int row  = lane & 15;
  const int kb   = (lane >> 4) << 3;               // 0 or 8
  const bf16_t* p = base + row * stride + kb;
  v8bf c0 = *(const v8bf*)(p);
  v8bf c1 = *(const v8bf*)(p + 16);
  v16bf a;
#pragma unroll
  for (int i = 0; i < 8; ++i) { a[i] = c0[i]; a[i + 8] = c1[i]; }
  return a;
}

__device__ __forceinline__ v16bf load_b_frag(const bf16_t* base, int stride) {
  const int lane = threadIdx.x & 31;
  const int n    = lane & 15;
  const int ko   = (lane >> 4) << 3;               // 0 or 8
  const bf16_t* p = base + n * stride + ko;
  v8bf c0 = *(const v8bf*)(p);
  v8bf c1 = *(const v8bf*)(p + 16);
  v16bf b;
#pragma unroll
  for (int i = 0; i < 8; ++i) { b[i] = c0[i]; b[i + 8] = c1[i]; }
  return b;
}

// ---------------------------------------------------------------------------
// Elementwise cast f32 -> bf16
// ---------------------------------------------------------------------------
__global__ __launch_bounds__(256)
void cast_f32_bf16(const float* __restrict__ in, bf16_t* __restrict__ out, int n) {
  int i = blockIdx.x * 256 + threadIdx.x;
  if (i < n) out[i] = (bf16_t)in[i];
}

// ---------------------------------------------------------------------------
// Transpose + cast: W [K][N] f32 row-major -> Wt [N][K] bf16
// ---------------------------------------------------------------------------
__global__ __launch_bounds__(256)
void transpose_cast(const float* __restrict__ W, bf16_t* __restrict__ Wt,
                    int K, int N) {
  __shared__ float tile[32][33];
  const int tx = threadIdx.x & 31;
  const int ty = threadIdx.x >> 5;                 // 0..7
  const int n0 = blockIdx.x * 32;
  const int k0 = blockIdx.y * 32;
#pragma unroll
  for (int i = 0; i < 32; i += 8)
    tile[ty + i][tx] = W[(size_t)(k0 + ty + i) * N + n0 + tx];
  __syncthreads();
#pragma unroll
  for (int i = 0; i < 32; i += 8)
    Wt[(size_t)(n0 + ty + i) * K + k0 + tx] = (bf16_t)tile[tx][ty + i];
}

// ---------------------------------------------------------------------------
// V [tok][1024] bf16 -> Vt [(b*16+h)*64+d][SEQ] bf16   (coalesced writes)
// ---------------------------------------------------------------------------
__global__ __launch_bounds__(256)
void v_transpose(const bf16_t* __restrict__ V, bf16_t* __restrict__ Vt) {
  size_t i = (size_t)blockIdx.x * 256 + threadIdx.x;   // over B*H*DK*SEQ
  int s = (int)(i & (SEQ - 1));
  size_t rest = i >> 11;
  int d = (int)(rest & (DK - 1)); rest >>= 6;
  int h = (int)(rest & (N_HEADS - 1));
  int b = (int)(rest >> 4);
  Vt[i] = V[(size_t)(b * SEQ + s) * D_MODEL + h * DK + d];
}

// ---------------------------------------------------------------------------
// GEMM: C[M][N] = epi(A[M][K] @ Bt[N][K]^T + bias[N] (+res)) -> f32 and/or bf16
// 128x128 block tile, 8 waves, wave = 4x2 frags of 16x16, K-step 64,
// double-buffered LDS fed by async global->LDS (ASYNCcnt) when available.
// ---------------------------------------------------------------------------
template<bool RELU, bool RES>
__global__ __launch_bounds__(256)
void gemm_bf16_wmma(const bf16_t* __restrict__ A, const bf16_t* __restrict__ Bt,
                    const float* __restrict__ bias, const float* __restrict__ res,
                    float* __restrict__ outF, bf16_t* __restrict__ outB,
                    int M, int N, int K) {
  constexpr int LDSK = 72;                          // 64 + 8 pad (bank-safe)
  __shared__ alignas(16) bf16_t sA[2][128 * LDSK];
  __shared__ alignas(16) bf16_t sB[2][128 * LDSK];

  const int tid  = threadIdx.x;
  const int lane = tid & 31;
  const int wid  = tid >> 5;
  const int wr   = wid >> 2;                        // 0..1
  const int wc   = wid & 3;                        // 0..3
  const int m0   = blockIdx.y * 128;
  const int n0   = blockIdx.x * 128;

  v8f acc[4][2];
#pragma unroll
  for (int mi = 0; mi < 4; ++mi)
#pragma unroll
    for (int ni = 0; ni < 2; ++ni)
#pragma unroll
      for (int r = 0; r < 8; ++r) acc[mi][ni][r] = 0.0f;

#if USE_ASYNC_LDS
  auto load_tile = [&](int buf, int k0) {
#pragma unroll
    for (int i = 0; i < 4; ++i) {                   // 1024 b128 chunks / matrix
      int idx = tid + i * 256;
      int row = idx >> 3;
      int c   = (idx & 7) * 8;
      async_copy_b128(&A[(size_t)(m0 + row) * K + k0 + c],
                      &sA[buf][row * LDSK + c]);
      async_copy_b128(&Bt[(size_t)(n0 + row) * K + k0 + c],
                      &sB[buf][row * LDSK + c]);
    }
  };
  load_tile(0, 0);
  __builtin_amdgcn_s_wait_asynccnt(0);
#else
  v8bf ra[4], rb[4];
  auto fetch_regs = [&](int k0) {
#pragma unroll
    for (int i = 0; i < 4; ++i) {
      int idx = tid + i * 256;
      int row = idx >> 3;
      int c   = (idx & 7) * 8;
      ra[i] = *(const v8bf*)&A[(size_t)(m0 + row) * K + k0 + c];
      rb[i] = *(const v8bf*)&Bt[(size_t)(n0 + row) * K + k0 + c];
    }
  };
  auto store_regs = [&](int buf) {
#pragma unroll
    for (int i = 0; i < 4; ++i) {
      int idx = tid + i * 256;
      int row = idx >> 3;
      int c   = (idx & 7) * 8;
      *(v8bf*)&sA[buf][row * LDSK + c] = ra[i];
      *(v8bf*)&sB[buf][row * LDSK + c] = rb[i];
    }
  };
  fetch_regs(0);
  store_regs(0);
#endif
  __syncthreads();

  const int NT = K >> 6;                            // K / 64
  for (int kt = 0; kt < NT; ++kt) {
    const int  cur  = kt & 1;
    const bool more = (kt + 1 < NT);
#if USE_ASYNC_LDS
    if (more) load_tile((kt + 1) & 1, (kt + 1) * 64);   // overlaps compute
#else
    if (more) fetch_regs((kt + 1) * 64);
#endif
#pragma unroll
    for (int kk = 0; kk < 2; ++kk) {
      v16bf af[4], bfr[2];
#pragma unroll
      for (int mi = 0; mi < 4; ++mi)
        af[mi] = load_a_frag(&sA[cur][(wr * 64 + mi * 16) * LDSK + kk * 32], LDSK);
#pragma unroll
      for (int ni = 0; ni < 2; ++ni)
        bfr[ni] = load_b_frag(&sB[cur][(wc * 32 + ni * 16) * LDSK + kk * 32], LDSK);
#pragma unroll
      for (int mi = 0; mi < 4; ++mi)
#pragma unroll
        for (int ni = 0; ni < 2; ++ni)
          acc[mi][ni] = __builtin_amdgcn_wmma_f32_16x16x32_bf16(
              false, af[mi], false, bfr[ni], (short)0, acc[mi][ni], false, false);
    }
#if USE_ASYNC_LDS
    if (more) __builtin_amdgcn_s_wait_asynccnt(0);
#else
    if (more) store_regs((kt + 1) & 1);
#endif
    __syncthreads();
  }

  // Epilogue. C layout: VGPR r -> row r + (lane>=16 ? 8:0); col = lane&15.
  const int hi8 = (lane >> 4) * 8;
  const int nlo = lane & 15;
#pragma unroll
  for (int mi = 0; mi < 4; ++mi) {
#pragma unroll
    for (int ni = 0; ni < 2; ++ni) {
      const int col = n0 + wc * 32 + ni * 16 + nlo;
      const float bv = bias[col];
#pragma unroll
      for (int r = 0; r < 8; ++r) {
        const int row = m0 + wr * 64 + mi * 16 + hi8 + r;
        float v = acc[mi][ni][r] + bv;
        if (RELU) v = fmaxf(v, 0.0f);
        if (RES)  v += res[(size_t)row * N + col];
        if (outF) outF[(size_t)row * N + col] = v;
        if (outB) outB[(size_t)row * N + col] = (bf16_t)v;
      }
    }
  }
}

// ---------------------------------------------------------------------------
// Flash attention: block = 64 queries (4 independent waves x 16 queries),
// grid.y = b*H + h. Streams K/V in 32-key tiles; online softmax; P routed
// through a per-wave padded LDS tile to re-shape C-frag -> A-frag.
// ---------------------------------------------------------------------------
__global__ __launch_bounds__(128)
void flash_attn_wmma(const bf16_t* __restrict__ Q, const bf16_t* __restrict__ Kmat,
                     const bf16_t* __restrict__ Vt, bf16_t* __restrict__ O) {
  __shared__ alignas(16) bf16_t pshare[4][16 * 40];
  const int tid  = threadIdx.x;
  const int lane = tid & 31;
  const int wid  = tid >> 5;                        // 0..3
  const int b    = blockIdx.y >> 4;
  const int h    = blockIdx.y & 15;
  const int q0   = blockIdx.x * 64 + wid * 16;

  const bf16_t* Qbase = Q    + (size_t)(b * SEQ + q0) * D_MODEL + h * DK;
  const bf16_t* Kbase = Kmat + (size_t)b * SEQ * D_MODEL + h * DK;
  const bf16_t* Vbase = Vt   + (size_t)(b * N_HEADS + h) * DK * SEQ;

  v16bf qf[2];
  qf[0] = load_a_frag(Qbase,      D_MODEL);         // d = 0..31
  qf[1] = load_a_frag(Qbase + 32, D_MODEL);         // d = 32..63

  v8f acc[4];
  float mrun[8], lrun[8];
#pragma unroll
  for (int nf = 0; nf < 4; ++nf)
#pragma unroll
    for (int r = 0; r < 8; ++r) acc[nf][r] = 0.0f;
#pragma unroll
  for (int r = 0; r < 8; ++r) { mrun[r] = -1e30f; lrun[r] = 0.0f; }

  bf16_t* pw = &pshare[wid][0];
  const int hi8 = (lane >> 4) * 8;
  const int nlo = lane & 15;

  for (int j = 0; j < SEQ; j += 32) {
    if (j + 32 < SEQ) {                             // hint next KV tile into L2
      __builtin_prefetch(Kbase + (size_t)(j + 32) * D_MODEL, 0, 1);
      __builtin_prefetch(Vbase + j + 32, 0, 1);
    }
    // S = Q @ K^T  (K row-major [seq][d] IS the required Bt layout)
    v8f s[2];
#pragma unroll
    for (int nf = 0; nf < 2; ++nf) {
#pragma unroll
      for (int r = 0; r < 8; ++r) s[nf][r] = 0.0f;
#pragma unroll
      for (int kk = 0; kk < 2; ++kk) {
        v16bf kfrag = load_b_frag(Kbase + (size_t)(j + nf * 16) * D_MODEL + kk * 32,
                                  D_MODEL);
        s[nf] = __builtin_amdgcn_wmma_f32_16x16x32_bf16(
            false, qf[kk], false, kfrag, (short)0, s[nf], false, false);
      }
    }
    // online softmax; row groups are the 16 lanes sharing bit4 of lane id
#pragma unroll
    for (int r = 0; r < 8; ++r) {
      float s0 = s[0][r] * 0.125f;                  // 1/sqrt(64)
      float s1 = s[1][r] * 0.125f;
      float t = fmaxf(s0, s1);
      t = fmaxf(t, __shfl_xor(t, 1, 32));
      t = fmaxf(t, __shfl_xor(t, 2, 32));
      t = fmaxf(t, __shfl_xor(t, 4, 32));
      t = fmaxf(t, __shfl_xor(t, 8, 32));
      float mnew  = fmaxf(mrun[r], t);
      float alpha = __expf(mrun[r] - mnew);
      float p0 = __expf(s0 - mnew);
      float p1 = __expf(s1 - mnew);
      float ps = p0 + p1;
      ps += __shfl_xor(ps, 1, 32);
      ps += __shfl_xor(ps, 2, 32);
      ps += __shfl_xor(ps, 4, 32);
      ps += __shfl_xor(ps, 8, 32);
      lrun[r] = lrun[r] * alpha + ps;
      mrun[r] = mnew;
#pragma unroll
      for (int nf = 0; nf < 4; ++nf) acc[nf][r] *= alpha;
      pw[(hi8 + r) * 40 + nlo]      = (bf16_t)p0;   // P tile, C-layout -> LDS
      pw[(hi8 + r) * 40 + 16 + nlo] = (bf16_t)p1;
    }
    __builtin_amdgcn_wave_barrier();                // LDS same-wave RAW (in-order)
    v16bf pa = load_a_frag(pw, 40);
#pragma unroll
    for (int nf = 0; nf < 4; ++nf) {
      v16bf vfrag = load_b_frag(Vbase + (size_t)(nf * 16) * SEQ + j, SEQ);
      acc[nf] = __builtin_amdgcn_wmma_f32_16x16x32_bf16(
          false, pa, false, vfrag, (short)0, acc[nf], false, false);
    }
    __builtin_amdgcn_wave_barrier();
  }

#pragma unroll
  for (int nf = 0; nf < 4; ++nf)
#pragma unroll
    for (int r = 0; r < 8; ++r) {
      int row = q0 + hi8 + r;
      int col = h * DK + nf * 16 + nlo;
      O[(size_t)(b * SEQ + row) * D_MODEL + col] = (bf16_t)(acc[nf][r] / lrun[r]);
    }
}

// ---------------------------------------------------------------------------
// Row LayerNorm over D_MODEL=1024, one block per row; emits f32 and/or bf16.
// ---------------------------------------------------------------------------
__global__ __launch_bounds__(256)
void layernorm_rows(const float* __restrict__ Y, const float* __restrict__ g,
                    const float* __restrict__ be, float* __restrict__ outF,
                    bf16_t* __restrict__ outB) {
  __shared__ float red[256];
  const int row = blockIdx.x;
  const int tid = threadIdx.x;
  const float* y = Y + (size_t)row * D_MODEL;
  float v[4];
  float s = 0.0f;
#pragma unroll
  for (int i = 0; i < 4; ++i) { v[i] = y[tid + i * 256]; s += v[i]; }
  red[tid] = s; __syncthreads();
  for (int o = 128; o > 0; o >>= 1) { if (tid < o) red[tid] += red[tid + o]; __syncthreads(); }
  const float mu = red[0] * (1.0f / D_MODEL);
  __syncthreads();
  float s2 = 0.0f;
#pragma unroll
  for (int i = 0; i < 4; ++i) { float d = v[i] - mu; s2 += d * d; }
  red[tid] = s2; __syncthreads();
  for (int o = 128; o > 0; o >>= 1) { if (tid < o) red[tid] += red[tid + o]; __syncthreads(); }
  const float rstd = rsqrtf(red[0] * (1.0f / D_MODEL) + 1e-5f);
#pragma unroll
  for (int i = 0; i < 4; ++i) {
    int c = tid + i * 256;
    float o = (v[i] - mu) * rstd * g[c] + be[c];
    if (outF) outF[(size_t)row * D_MODEL + c] = o;
    if (outB) outB[(size_t)row * D_MODEL + c] = (bf16_t)o;
  }
}

// ---------------------------------------------------------------------------
// Host orchestration
// ---------------------------------------------------------------------------
extern "C" void kernel_launch(void* const* d_in, const int* in_sizes, int n_in,
                              void* d_out, int out_size, void* d_ws, size_t ws_size,
                              hipStream_t stream) {
  (void)in_sizes; (void)n_in; (void)out_size; (void)ws_size;
  const float* x   = (const float*)d_in[0];
  const float* Wq  = (const float*)d_in[1];
  const float* bq  = (const float*)d_in[2];
  const float* Wk  = (const float*)d_in[3];
  const float* bk  = (const float*)d_in[4];
  const float* Wv  = (const float*)d_in[5];
  const float* bv  = (const float*)d_in[6];
  const float* Wo  = (const float*)d_in[7];
  const float* bo  = (const float*)d_in[8];
  const float* W1  = (const float*)d_in[9];
  const float* b1  = (const float*)d_in[10];
  const float* W2  = (const float*)d_in[11];
  const float* b2  = (const float*)d_in[12];
  const float* g1  = (const float*)d_in[13];
  const float* be1 = (const float*)d_in[14];
  const float* g2  = (const float*)d_in[15];
  const float* be2 = (const float*)d_in[16];
  float* out = (float*)d_out;

  char* ws = (char*)d_ws;
  size_t off = 0;
  auto alloc = [&](size_t bytes) -> void* {
    off = (off + 255) & ~(size_t)255;
    void* p = ws + off;
    off += bytes;
    return p;
  };

  const size_t TD = (size_t)TOKENS * D_MODEL;
  bf16_t* xb   = (bf16_t*)alloc(TD * 2);
  bf16_t* Wqt  = (bf16_t*)alloc((size_t)D_MODEL * D_MODEL * 2);
  bf16_t* Wkt  = (bf16_t*)alloc((size_t)D_MODEL * D_MODEL * 2);
  bf16_t* Wvt  = (bf16_t*)alloc((size_t)D_MODEL * D_MODEL * 2);
  bf16_t* Wot  = (bf16_t*)alloc((size_t)D_MODEL * D_MODEL * 2);
  bf16_t* W1t  = (bf16_t*)alloc((size_t)D_MODEL * D_FF * 2);
  bf16_t* W2t  = (bf16_t*)alloc((size_t)D_FF * D_MODEL * 2);
  bf16_t* Qb   = (bf16_t*)alloc(TD * 2);
  bf16_t* Kb   = (bf16_t*)alloc(TD * 2);
  bf16_t* Vb   = (bf16_t*)alloc(TD * 2);
  bf16_t* Vtb  = (bf16_t*)alloc(TD * 2);
  bf16_t* Ob   = (bf16_t*)alloc(TD * 2);
  float*  y1   = (float*)alloc(TD * 4);
  float*  hF   = (float*)alloc(TD * 4);
  bf16_t* hB   = (bf16_t*)alloc(TD * 2);
  bf16_t* ff1  = (bf16_t*)alloc((size_t)TOKENS * D_FF * 2);
  float*  y2   = (float*)alloc(TD * 4);

  // 1) precision conversion / weight re-layout
  cast_f32_bf16<<<(int)(TD / 256), 256, 0, stream>>>(x, xb, (int)TD);
  transpose_cast<<<dim3(D_MODEL / 32, D_MODEL / 32), 256, 0, stream>>>(Wq, Wqt, D_MODEL, D_MODEL);
  transpose_cast<<<dim3(D_MODEL / 32, D_MODEL / 32), 256, 0, stream>>>(Wk, Wkt, D_MODEL, D_MODEL);
  transpose_cast<<<dim3(D_MODEL / 32, D_MODEL / 32), 256, 0, stream>>>(Wv, Wvt, D_MODEL, D_MODEL);
  transpose_cast<<<dim3(D_MODEL / 32, D_MODEL / 32), 256, 0, stream>>>(Wo, Wot, D_MODEL, D_MODEL);
  transpose_cast<<<dim3(D_FF / 32,    D_MODEL / 32), 256, 0, stream>>>(W1, W1t, D_MODEL, D_FF);
  transpose_cast<<<dim3(D_MODEL / 32, D_FF / 32),    256, 0, stream>>>(W2, W2t, D_FF, D_MODEL);

  // 2) QKV projections
  gemm_bf16_wmma<false, false><<<dim3(D_MODEL / 128, TOKENS / 128), 256, 0, stream>>>(
      xb, Wqt, bq, nullptr, nullptr, Qb, TOKENS, D_MODEL, D_MODEL);
  gemm_bf16_wmma<false, false><<<dim3(D_MODEL / 128, TOKENS / 128), 256, 0, stream>>>(
      xb, Wkt, bk, nullptr, nullptr, Kb, TOKENS, D_MODEL, D_MODEL);
  gemm_bf16_wmma<false, false><<<dim3(D_MODEL / 128, TOKENS / 128), 256, 0, stream>>>(
      xb, Wvt, bv, nullptr, nullptr, Vb, TOKENS, D_MODEL, D_MODEL);
  v_transpose<<<(int)(TD / 256), 256, 0, stream>>>(Vb, Vtb);

  // 3) attention
  flash_attn_wmma<<<dim3(SEQ / 64, BATCH * N_HEADS), 128, 0, stream>>>(Qb, Kb, Vtb, Ob);

  // 4) output projection + residual, then LN1
  gemm_bf16_wmma<false, true><<<dim3(D_MODEL / 128, TOKENS / 128), 256, 0, stream>>>(
      Ob, Wot, bo, x, y1, nullptr, TOKENS, D_MODEL, D_MODEL);
  layernorm_rows<<<TOKENS, 256, 0, stream>>>(y1, g1, be1, hF, hB);

  // 5) FFN
  gemm_bf16_wmma<true, false><<<dim3(D_FF / 128, TOKENS / 128), 256, 0, stream>>>(
      hB, W1t, b1, nullptr, nullptr, ff1, TOKENS, D_FF, D_MODEL);
  gemm_bf16_wmma<false, true><<<dim3(D_MODEL / 128, TOKENS / 128), 256, 0, stream>>>(
      ff1, W2t, b2, hF, y2, nullptr, TOKENS, D_MODEL, D_FF);

  // 6) LN2 -> final f32 output
  layernorm_rows<<<TOKENS, 256, 0, stream>>>(y2, g2, be2, out, nullptr);
}
